// Llama3_85066122264688
// MI455X (gfx1250) — matrix operand order repaired and checked by reading, not compiled, not run on hardware
//
// Llama3 2-layer forward for MI455X (gfx1250), bf16-WMMA everywhere, fp32 accum.
// Compute-bound (~630 GFLOP vs ~60us of HBM traffic at 23.3 TB/s), so all
// GEMMs + attention matmuls use v_wmma_f32_16x16x32_bf16 (8x the FLOP/instr of
// the f32 WMMA path). Weights are pre-converted fp32->bf16 by a tiny streaming
// kernel right before each GEMM (bandwidth-trivial on this chip), so the GEMM
// hot loop is pure async global->LDS (ASYNCcnt) + ds_load_b128 + WMMA with a
// double-buffered LDS pipeline. Residual stream stays fp32.
// Requires ~361MB workspace.
#include <hip/hip_runtime.h>
#include <hip/hip_bf16.h>
#include <math.h>

#define TT   2048
#define DD   2048
#define FFN  8192
#define NH   32
#define NKV  8
#define DHD  64
#define GQA  4
#define NL   2
#define NV   32000
#define EPSF 1e-5f

typedef __bf16 bf16_t;
typedef bf16_t v16bf __attribute__((ext_vector_type(16)));
typedef float  v8f   __attribute__((ext_vector_type(8)));
typedef int    v4i   __attribute__((vector_size(4 * sizeof(int))));

union FragBF { v16bf v; uint4 u[2]; };

__device__ __forceinline__ unsigned short f2bf(float f) {
  unsigned int u = __builtin_bit_cast(unsigned int, f);
  u += 0x7FFFu + ((u >> 16) & 1u);           // round-to-nearest-even
  return (unsigned short)(u >> 16);
}
__device__ __forceinline__ float bf2f(unsigned short h) {
  unsigned int u = ((unsigned int)h) << 16;
  return __builtin_bit_cast(float, u);
}
__device__ __forceinline__ unsigned int pk2bf(float x, float y) {
  return (unsigned int)f2bf(x) | ((unsigned int)f2bf(y) << 16);
}

__device__ __forceinline__ v8f wmma_bf16(v16bf a, v16bf b, v8f c) {
  return __builtin_amdgcn_wmma_f32_16x16x32_bf16(false, a, false, b, (short)0, c,
                                                 false, false);
}

// ---- gfx1250 async global->LDS (ASYNCcnt) with safe fallbacks ----
#if defined(__has_builtin)
#if __has_builtin(__builtin_amdgcn_global_load_async_to_lds_b128)
#define HAVE_ASYNC_LDS 1
#endif
#endif

__device__ __forceinline__ void async_g2l_b128(const void* g, void* l) {
#if defined(HAVE_ASYNC_LDS)
  __builtin_amdgcn_global_load_async_to_lds_b128(
      (__attribute__((address_space(1))) v4i*)g,
      (__attribute__((address_space(3))) v4i*)l, 0, 0);
#else
  *(uint4*)l = *(const uint4*)g;             // synchronous fallback
#endif
}

__device__ __forceinline__ void wait_async_all() {
#if defined(HAVE_ASYNC_LDS)
#if __has_builtin(__builtin_amdgcn_s_wait_asynccnt)
  __builtin_amdgcn_s_wait_asynccnt(0);
#else
  asm volatile("s_wait_asynccnt 0x0" ::: "memory");
#endif
#endif
}

// ---------------- elementwise kernels ----------------

__global__ void k_gather(const int* __restrict__ tok, const float* __restrict__ emb,
                         float* __restrict__ x) {
  int i = blockIdx.x * blockDim.x + threadIdx.x;      // over T*D
  int t = i / DD, d = i % DD;
  x[i] = emb[(size_t)tok[t] * DD + d];
}

// streaming fp32 -> bf16 (4 elements/thread); n4 = n/4
__global__ void k_cvt_bf16(const float* __restrict__ in,
                           unsigned short* __restrict__ out, long long n4) {
  long long i = (long long)blockIdx.x * blockDim.x + threadIdx.x;
  if (i >= n4) return;
  float4 f = ((const float4*)in)[i];
  unsigned long long pk = (unsigned long long)pk2bf(f.x, f.y) |
                          ((unsigned long long)pk2bf(f.z, f.w) << 32);
  ((unsigned long long*)out)[i] = pk;
}

__global__ void k_rope_tab(float* __restrict__ cosb, float* __restrict__ sinb) {
  int i = blockIdx.x * blockDim.x + threadIdx.x;      // over T*DH
  if (i >= TT * DHD) return;
  int t = i / DHD, d = i % DHD;
  int fi = d % (DHD / 2);
  float j = (float)(2 * fi) / (float)DHD;
  float inv = powf(500000.0f, -j);
  float wl = 2.0f * 3.14159265358979323846f / inv;
  const float factor = 32.f, hi = 4.f, lo = 1.f, orig = 8192.f;
  const float cut0 = orig / hi, cut1 = orig / lo;     // (2048, 8192)
  float inv2 = (wl > cut1) ? inv / factor : inv;
  float smooth = (orig / wl - lo) / (hi - lo);
  float smoothed = (1.f - smooth) * inv / factor + smooth * inv;
  bool med = (!(wl < cut1)) && (!(wl > cut0));        // faithful to reference
  float f = med ? smoothed : inv2;
  float a = (float)t * f;
  cosb[i] = cosf(a);
  sinb[i] = sinf(a);
}

__global__ void k_rmsnorm(const float* __restrict__ x, const float* __restrict__ w,
                          unsigned short* __restrict__ y) {
  __shared__ float red[256];
  int t = blockIdx.x;
  const float* xr = x + (size_t)t * DD;
  float s = 0.f;
  for (int d = threadIdx.x; d < DD; d += 256) { float v = xr[d]; s += v * v; }
  red[threadIdx.x] = s;
  __syncthreads();
  for (int off = 128; off; off >>= 1) {
    if ((int)threadIdx.x < off) red[threadIdx.x] += red[threadIdx.x + off];
    __syncthreads();
  }
  float r = rsqrtf(red[0] / (float)DD + EPSF);
  for (int d = threadIdx.x; d < DD; d += 256)
    y[(size_t)t * DD + d] = f2bf(xr[d] * r * w[d]);
}

// rope + cvt + transpose: in [T, nheads*DH] fp32 -> out [nheads, T, DH] bf16
__global__ void k_rope_cvt(const float* __restrict__ in, const float* __restrict__ cosb,
                           const float* __restrict__ sinb, unsigned short* __restrict__ out,
                           int nheads) {
  int i = blockIdx.x * blockDim.x + threadIdx.x;
  if (i >= TT * nheads * DHD) return;
  int d = i % DHD;
  int h = (i / DHD) % nheads;
  int t = i / (DHD * nheads);
  const float* row = in + (size_t)t * nheads * DHD + (size_t)h * DHD;
  float v = row[d];
  float o = (d < DHD / 2) ? -row[d + DHD / 2] : row[d - DHD / 2];
  float r = v * cosb[t * DHD + d] + o * sinb[t * DHD + d];
  out[((size_t)h * TT + t) * DHD + d] = f2bf(r);
}

// cvt + transpose: v [T, NKV*DH] fp32 -> vT [NKV, DH, T] bf16
__global__ void k_vT(const float* __restrict__ v, unsigned short* __restrict__ vT) {
  int i = blockIdx.x * blockDim.x + threadIdx.x;
  if (i >= TT * NKV * DHD) return;
  int d = i % DHD;
  int h = (i / DHD) % NKV;
  int t = i / (DHD * NKV);
  vT[((size_t)h * DHD + d) * TT + t] = f2bf(v[(size_t)t * NKV * DHD + h * DHD + d]);
}

__global__ void k_silu_mul(const float* __restrict__ up, const float* __restrict__ gate,
                           unsigned short* __restrict__ hb) {
  int i = blockIdx.x * blockDim.x + threadIdx.x;      // over T*FF
  float g = gate[i];
  float sg = g / (1.f + expf(-g));
  hb[i] = f2bf(up[i] * sg);
}

// ---------------- NT GEMM: C[M,N] = A_bf16[M,K] . B_bf16[N,K]^T (+Cadd) ----------------
// Both operands bf16 in global; tiles move via async global->LDS B128 copies
// (ASYNCcnt), double-buffered so tile k+1 is in flight during tile k's WMMAs.
// One barrier per K-step; inner loop is async-copy + ds_load_b128 + WMMA only.

#define BM 128
#define BN 128
#define BK 32
#define LDT 48  // padded LDS row stride in halves (96B, 16B-aligned chunks)

__global__ __launch_bounds__(256) void k_gemm_nt(const unsigned short* __restrict__ A,
                                                 const unsigned short* __restrict__ Bw,
                                                 const float* __restrict__ Cadd,
                                                 float* __restrict__ Cout,
                                                 int M, int N, int K) {
  __shared__ unsigned short As[2][BM * LDT];
  __shared__ unsigned short Bs[2][BN * LDT];
  int tid = threadIdx.x;
  int wave = tid >> 5, lane = tid & 31;
  int wm = wave >> 2, wn = wave & 3;                  // 2x4 wave grid
  int lhalf = lane >> 4, lmod = lane & 15;
  int m0 = blockIdx.y * BM;
  int n0 = blockIdx.x * BN;

  // per-thread copy coordinates: 2 x 16B chunks per tile per operand
  int arow = tid >> 2, acol = (tid & 3) * 8;          // rows arow, arow+64
  const unsigned short* Ag0 = A + (size_t)(m0 + arow) * K + acol;
  const unsigned short* Ag1 = Ag0 + (size_t)64 * K;
  const unsigned short* Bg0 = Bw + (size_t)(n0 + arow) * K + acol;
  const unsigned short* Bg1 = Bg0 + (size_t)64 * K;
  unsigned short* Al0[2] = {&As[0][arow * LDT + acol], &As[1][arow * LDT + acol]};
  unsigned short* Al1[2] = {&As[0][(arow + 64) * LDT + acol], &As[1][(arow + 64) * LDT + acol]};
  unsigned short* Bl0[2] = {&Bs[0][arow * LDT + acol], &Bs[1][arow * LDT + acol]};
  unsigned short* Bl1[2] = {&Bs[0][(arow + 64) * LDT + acol], &Bs[1][(arow + 64) * LDT + acol]};

  v8f zero = {0.f, 0.f, 0.f, 0.f, 0.f, 0.f, 0.f, 0.f};
  v8f acc[4][2];
  for (int i = 0; i < 4; ++i)
    for (int j = 0; j < 2; ++j) acc[i][j] = zero;

  int nk = K / BK;

  // prologue: tile 0 in flight
  async_g2l_b128(Ag0, Al0[0]);
  async_g2l_b128(Ag1, Al1[0]);
  async_g2l_b128(Bg0, Bl0[0]);
  async_g2l_b128(Bg1, Bl1[0]);

  for (int ki = 0; ki < nk; ++ki) {
    int cur = ki & 1;
    wait_async_all();                                 // tile `cur` landed in LDS
    __syncthreads();

    // kick off tile ki+1 while we compute tile ki
    if (ki + 1 < nk) {
      int kb = (ki + 1) * BK;
      int nxt = cur ^ 1;
      async_g2l_b128(Ag0 + kb, Al0[nxt]);
      async_g2l_b128(Ag1 + kb, Al1[nxt]);
      async_g2l_b128(Bg0 + kb, Bl0[nxt]);
      async_g2l_b128(Bg1 + kb, Bl1[nxt]);
    }

    FragBF afr[4], bfr[2];
#pragma unroll
    for (int i = 0; i < 4; ++i) {                     // A: 16x32, M=lane%16
      int r = wm * 64 + i * 16 + lmod;
      afr[i].u[0] = *(const uint4*)&As[cur][r * LDT + lhalf * 8];
      afr[i].u[1] = *(const uint4*)&As[cur][r * LDT + 16 + lhalf * 8];
    }
#pragma unroll
    for (int j = 0; j < 2; ++j) {                     // B: 32x16, N=lane%16
      int c = wn * 32 + j * 16 + lmod;
      bfr[j].u[0] = *(const uint4*)&Bs[cur][c * LDT + lhalf * 8];
      bfr[j].u[1] = *(const uint4*)&Bs[cur][c * LDT + 16 + lhalf * 8];
    }
#pragma unroll
    for (int i = 0; i < 4; ++i)
#pragma unroll
      for (int j = 0; j < 2; ++j)
        acc[i][j] = wmma_bf16(afr[i].v, bfr[j].v, acc[i][j]);
  }

  // writeback: row pointers stepped by N, residual branch hoisted per tile
  int row0 = m0 + wm * 64 + 8 * lhalf;                // C: M = r + 8*(lane/16)
  int col0 = n0 + wn * 32 + lmod;
#pragma unroll
  for (int i = 0; i < 4; ++i)
#pragma unroll
    for (int j = 0; j < 2; ++j) {
      size_t base = (size_t)(row0 + i * 16) * N + (col0 + j * 16);
      float* op = Cout + base;
      if (Cadd) {
        const float* ap = Cadd + base;
#pragma unroll
        for (int r = 0; r < 8; ++r)
          op[(size_t)r * N] = acc[i][j][r] + ap[(size_t)r * N];
      } else {
#pragma unroll
        for (int r = 0; r < 8; ++r)
          op[(size_t)r * N] = acc[i][j][r];
      }
    }
}

// ---------------- attention: one block per (head, 16-row tile) ----------------

__global__ __launch_bounds__(128) void k_attn(const unsigned short* __restrict__ qb,  // [H,T,DH]
                                              const unsigned short* __restrict__ kb,  // [NKV,T,DH]
                                              const unsigned short* __restrict__ vT,  // [NKV,DH,T]
                                              unsigned short* __restrict__ outb) {    // [T,D]
  __shared__ unsigned short S[16 * TT];   // score/prob strip (bf16), 64KB
  __shared__ float red[16 * 8];
  __shared__ float rinv[16];
  int h = blockIdx.y, t0 = blockIdx.x * 16;
  int hkv = h / GQA;
  int tid = threadIdx.x, wave = tid >> 5, lane = tid & 31;
  int lhalf = lane >> 4, lmod = lane & 15;

  const unsigned short* Q  = qb + ((size_t)h * TT + t0) * DHD;
  const unsigned short* Kp = kb + (size_t)hkv * TT * DHD;
  const unsigned short* Vp = vT + (size_t)hkv * DHD * TT;

  // Q fragments once (A 16x32 layout), two K-steps over DH=64
  FragBF qf[2];
#pragma unroll
  for (int ks = 0; ks < 2; ++ks) {
    const unsigned short* base = Q + (size_t)lmod * DHD + ks * 32;
    qf[ks].u[0] = *(const uint4*)&base[lhalf * 8];
    qf[ks].u[1] = *(const uint4*)&base[16 + lhalf * 8];
  }

  const float scale = 0.125f;  // 1/sqrt(64)
  for (int nt = wave; nt < TT / 16; nt += 4) {
    int s0 = nt * 16;
    v8f c = {0.f, 0.f, 0.f, 0.f, 0.f, 0.f, 0.f, 0.f};
#pragma unroll
    for (int ks = 0; ks < 2; ++ks) {
      FragBF bf;  // B 32x16: N = key index, K = d (contiguous in memory)
      const unsigned short* kbase = Kp + (size_t)(s0 + lmod) * DHD + ks * 32;
      bf.u[0] = *(const uint4*)&kbase[lhalf * 8];
      bf.u[1] = *(const uint4*)&kbase[16 + lhalf * 8];
      c = wmma_bf16(qf[ks].v, bf.v, c);
    }
#pragma unroll
    for (int r = 0; r < 8; ++r) {
      int rl = r + 8 * lhalf;
      int scol = s0 + lmod;
      float v = (scol <= t0 + rl) ? c[r] * scale : -3.0e38f;
      S[rl * TT + scol] = f2bf(v);
    }
  }
  __syncthreads();

  // softmax: 8 threads per row, each owns 256 columns
  int r = tid >> 3, j = tid & 7;
  float m = -3.0e38f;
  for (int c = j * 256; c < j * 256 + 256; ++c) m = fmaxf(m, bf2f(S[r * TT + c]));
  red[r * 8 + j] = m;
  __syncthreads();
  if (j == 0) {
    float mm = red[r * 8];
    for (int q = 1; q < 8; ++q) mm = fmaxf(mm, red[r * 8 + q]);
    red[r * 8] = mm;
  }
  __syncthreads();
  float rowm = red[r * 8];
  __syncthreads();
  float ssum = 0.f;
  for (int c = j * 256; c < j * 256 + 256; ++c) {
    float e = expf(bf2f(S[r * TT + c]) - rowm);
    ssum += e;
    S[r * TT + c] = f2bf(e);                 // unnormalized probs, bf16
  }
  red[r * 8 + j] = ssum;
  __syncthreads();
  if (j == 0) {
    float t = 0.f;
    for (int q = 0; q < 8; ++q) t += red[r * 8 + q];
    rinv[r] = 1.f / t;
  }
  __syncthreads();

  // ctx: each wave owns one 16-wide slice of DH; K-loop over all keys
  int d0 = wave * 16;
  v8f o = {0.f, 0.f, 0.f, 0.f, 0.f, 0.f, 0.f, 0.f};
  for (int sb = 0; sb < TT; sb += 32) {
    FragBF af, bf;
    const unsigned short* pr = &S[lmod * TT + sb];            // A: P strip in LDS
    af.u[0] = *(const uint4*)&pr[lhalf * 8];
    af.u[1] = *(const uint4*)&pr[16 + lhalf * 8];
    const unsigned short* vb = Vp + (size_t)(d0 + lmod) * TT + sb;  // B: V^T rows
    bf.u[0] = *(const uint4*)&vb[lhalf * 8];
    bf.u[1] = *(const uint4*)&vb[16 + lhalf * 8];
    o = wmma_bf16(af.v, bf.v, o);
  }
#pragma unroll
  for (int r2 = 0; r2 < 8; ++r2) {
    int rl = r2 + 8 * lhalf;
    outb[(size_t)(t0 + rl) * DD + h * DHD + d0 + lmod] = f2bf(o[r2] * rinv[rl]);
  }
}

// ---------------- host orchestration ----------------

static inline void launch_gemm(const unsigned short* A, const unsigned short* Bw,
                               const float* Cadd, float* Cout,
                               int M, int N, int K, hipStream_t s) {
  dim3 g(N / BN, M / BM);
  k_gemm_nt<<<g, 256, 0, s>>>(A, Bw, Cadd, Cout, M, N, K);
}

extern "C" void kernel_launch(void* const* d_in, const int* in_sizes, int n_in,
                              void* d_out, int out_size, void* d_ws, size_t ws_size,
                              hipStream_t stream) {
  (void)in_sizes; (void)n_in; (void)out_size; (void)ws_size;
  const int*   tok    = (const int*)d_in[0];
  const float* emb    = (const float*)d_in[1];
  const float* Wq     = (const float*)d_in[2];
  const float* Wk     = (const float*)d_in[3];
  const float* Wv     = (const float*)d_in[4];
  const float* Wo     = (const float*)d_in[5];
  const float* Wup    = (const float*)d_in[6];
  const float* Wgate  = (const float*)d_in[7];
  const float* Wdown  = (const float*)d_in[8];
  const float* n1     = (const float*)d_in[9];
  const float* n2     = (const float*)d_in[10];
  const float* norm_f = (const float*)d_in[11];

  char* base = (char*)d_ws;
  size_t off = 0;
  auto take = [&](size_t bytes) -> char* {
    char* p = base + off;
    off += (bytes + 255) & ~(size_t)255;
    return p;
  };
  float*          x     = (float*)take((size_t)TT * DD * 4);
  unsigned short* xb    = (unsigned short*)take((size_t)TT * DD * 2);
  float*          q     = (float*)take((size_t)TT * DD * 4);
  float*          kk    = (float*)take((size_t)TT * NKV * DHD * 4);
  float*          vv    = (float*)take((size_t)TT * NKV * DHD * 4);
  unsigned short* qbuf  = (unsigned short*)take((size_t)NH * TT * DHD * 2);
  unsigned short* kbuf  = (unsigned short*)take((size_t)NKV * TT * DHD * 2);
  unsigned short* vTb   = (unsigned short*)take((size_t)NKV * DHD * TT * 2);
  unsigned short* attn  = (unsigned short*)take((size_t)TT * DD * 2);
  float*          up    = (float*)take((size_t)TT * FFN * 4);
  float*          gate  = (float*)take((size_t)TT * FFN * 4);
  unsigned short* hb    = (unsigned short*)take((size_t)TT * FFN * 2);
  float*          cosb  = (float*)take((size_t)TT * DHD * 4);
  float*          sinb  = (float*)take((size_t)TT * DHD * 4);
  unsigned short* wbuf  = (unsigned short*)take((size_t)NV * DD * 2);  // bf16 weight staging

  auto cvtW = [&](const float* w, size_t n) {
    long long n4 = (long long)(n / 4);
    k_cvt_bf16<<<(unsigned)((n4 + 255) / 256), 256, 0, stream>>>(w, wbuf, n4);
  };

  k_gather<<<(TT * DD) / 256, 256, 0, stream>>>(tok, emb, x);
  k_rope_tab<<<(TT * DHD + 255) / 256, 256, 0, stream>>>(cosb, sinb);

  for (int l = 0; l < NL; ++l) {
    const float* wq = Wq + (size_t)l * DD * DD;
    const float* wk = Wk + (size_t)l * NKV * DHD * DD;
    const float* wv = Wv + (size_t)l * NKV * DHD * DD;
    const float* wo = Wo + (size_t)l * DD * DD;
    const float* wu = Wup + (size_t)l * FFN * DD;
    const float* wg = Wgate + (size_t)l * FFN * DD;
    const float* wd = Wdown + (size_t)l * DD * FFN;

    k_rmsnorm<<<TT, 256, 0, stream>>>(x, n1 + (size_t)l * DD, xb);
    cvtW(wq, (size_t)DD * DD);
    launch_gemm(xb, wbuf, nullptr, q, TT, DD, DD, stream);
    cvtW(wk, (size_t)NKV * DHD * DD);
    launch_gemm(xb, wbuf, nullptr, kk, TT, NKV * DHD, DD, stream);
    cvtW(wv, (size_t)NKV * DHD * DD);
    launch_gemm(xb, wbuf, nullptr, vv, TT, NKV * DHD, DD, stream);

    k_rope_cvt<<<(TT * NH * DHD) / 256, 256, 0, stream>>>(q, cosb, sinb, qbuf, NH);
    k_rope_cvt<<<(TT * NKV * DHD) / 256, 256, 0, stream>>>(kk, cosb, sinb, kbuf, NKV);
    k_vT<<<(TT * NKV * DHD) / 256, 256, 0, stream>>>(vv, vTb);

    k_attn<<<dim3(TT / 16, NH), 128, 0, stream>>>(qbuf, kbuf, vTb, attn);
    cvtW(wo, (size_t)DD * DD);
    launch_gemm(attn, wbuf, x, x, TT, DD, DD, stream);   // x += attn @ Wo^T

    k_rmsnorm<<<TT, 256, 0, stream>>>(x, n2 + (size_t)l * DD, xb);
    cvtW(wu, (size_t)FFN * DD);
    launch_gemm(xb, wbuf, nullptr, up, TT, FFN, DD, stream);
    cvtW(wg, (size_t)FFN * DD);
    launch_gemm(xb, wbuf, nullptr, gate, TT, FFN, DD, stream);
    k_silu_mul<<<(TT * FFN) / 256, 256, 0, stream>>>(up, gate, hb);
    cvtW(wd, (size_t)DD * FFN);
    launch_gemm(hb, wbuf, x, x, TT, DD, FFN, stream);    // x += (up*silu(gate)) @ Wdown^T
  }

  k_rmsnorm<<<TT, 256, 0, stream>>>(x, norm_f, xb);
  cvtW(emb, (size_t)NV * DD);
  launch_gemm(xb, wbuf, nullptr, (float*)d_out, TT, NV, DD, stream);  // tied head
}
